// Transformer1_47596827574508
// MI455X (gfx1250) — matrix operand order repaired and checked by reading, not compiled, not run on hardware
//
#include <hip/hip_runtime.h>
#include <hip/hip_bf16.h>
#include <cstdint>
#include <cstddef>

// ---------------------------------------------------------------------------
// Types / helpers
// ---------------------------------------------------------------------------
typedef __attribute__((ext_vector_type(16))) __bf16 bf16x16;
typedef __attribute__((ext_vector_type(8)))  float  f32x8;

struct BfFrag { union { bf16x16 v; unsigned int u[8]; }; };

__device__ inline __bf16 f2bf(float f) {
  unsigned int u = __builtin_bit_cast(unsigned int, f);
  u += 0x7FFFu + ((u >> 16) & 1u);           // round-to-nearest-even
  unsigned short s = (unsigned short)(u >> 16);
  return __builtin_bit_cast(__bf16, s);
}

__device__ inline f32x8 wmma_bf16(const BfFrag& a, const BfFrag& b, f32x8 c) {
  // D = A(16x32 bf16) x B(32x16 bf16) + C(16x16 f32)
  return __builtin_amdgcn_wmma_f32_16x16x32_bf16(
      /*neg_a=*/false, a.v, /*neg_b=*/false, b.v,
      /*c_mod=*/(short)0, c, /*reuse_a=*/false, /*reuse_b=*/false);
}

// A fragment (16x32) from a row-major tile with row stride 32 bf16.
// Lanes 0-15: row=lane, K in {0..7,16..23}; lanes 16-31: row=lane-16, K in {8..15,24..31}.
__device__ inline void load_a_frag(BfFrag& f, const __bf16* base, int lane) {
  const __bf16* p = base + (lane & 15) * 32;
  const int kb = (lane & 16) ? 8 : 0;
#pragma unroll
  for (int i = 0; i < 8; i++) {
    const int k = kb + (i < 4 ? 2 * i : 16 + 2 * (i - 4));
    f.u[i] = *reinterpret_cast<const unsigned int*>(p + k);
  }
}

// B fragment (32x16): source is an N-major (transposed) tile [n][k], row stride 32.
// Lanes 0-15: col=lane, K in {0..7,16..23}; lanes 16-31: col=lane-16, K in {8..15,24..31}.
__device__ inline void load_b_frag(BfFrag& f, const __bf16* base, int lane) {
  const __bf16* p = base + (lane & 15) * 32;
  const int kb = (lane & 16) ? 8 : 0;
#pragma unroll
  for (int i = 0; i < 8; i++)
    f.u[i] = *reinterpret_cast<const unsigned int*>(p + kb + (i < 4 ? 2 * i : 16 + 2 * (i - 4)));
}

// Async copy 16 bytes global -> LDS (GLOBAL_LOAD_ASYNC_TO_LDS_B128, ASYNCcnt).
__device__ inline void async_copy_b128(const void* gptr, void* lptr) {
  const unsigned lds = (unsigned)(uintptr_t)lptr;                 // low 32b = LDS addr
  const unsigned long long ga = (unsigned long long)(uintptr_t)gptr;
  asm volatile("global_load_async_to_lds_b128 %0, %1, off" :: "v"(lds), "v"(ga) : "memory");
}
__device__ inline void wait_async0() {
  asm volatile("s_wait_asynccnt 0" ::: "memory");
}

// ---------------------------------------------------------------------------
// fp32 -> bf16 weight conversion
// ---------------------------------------------------------------------------
__global__ __launch_bounds__(256) void f32_to_bf16_kernel(const float* __restrict__ in,
                                                          __bf16* __restrict__ out, int n) {
  int i = blockIdx.x * 256 + threadIdx.x;
  if (i < n) out[i] = f2bf(in[i]);
}

// ---------------------------------------------------------------------------
// Embedding: x[b,t,:] = emb[idx[b,t],:] + pos[t,:]
// ---------------------------------------------------------------------------
__global__ __launch_bounds__(256) void embed_kernel(const int* __restrict__ idx,
                                                    const float* __restrict__ emb,
                                                    const float* __restrict__ pos,
                                                    float* __restrict__ x) {
  const int token = blockIdx.x;            // 0..B*T-1
  const int t = token & 1023;
  const int e = threadIdx.x;
  const int id = idx[token];
  x[(size_t)token * 256 + e] = emb[(size_t)id * 256 + e] + pos[(size_t)t * 256 + e];
}

// ---------------------------------------------------------------------------
// LayerNorm over E=256, one 256-thread block per row. OUT_BF16 controls output.
// ---------------------------------------------------------------------------
template <bool OUT_BF16>
__global__ __launch_bounds__(256) void layernorm_kernel(const float* __restrict__ X,
                                                        const float* __restrict__ w,
                                                        const float* __restrict__ b,
                                                        void* __restrict__ out) {
  __shared__ float red[256];
  __shared__ float red2[256];
  const int row = blockIdx.x, tid = threadIdx.x;
  const float v = X[(size_t)row * 256 + tid];
  red[tid] = v;
  red2[tid] = v * v;
  __syncthreads();
  for (int s = 128; s > 0; s >>= 1) {
    if (tid < s) { red[tid] += red[tid + s]; red2[tid] += red2[tid + s]; }
    __syncthreads();
  }
  const float mean = red[0] * (1.0f / 256.0f);
  const float var  = red2[0] * (1.0f / 256.0f) - mean * mean;
  const float y = (v - mean) * rsqrtf(var + 1e-5f) * w[tid] + b[tid];
  if (OUT_BF16) ((__bf16*)out)[(size_t)row * 256 + tid] = f2bf(y);
  else          ((float*)out)[(size_t)row * 256 + tid] = y;
}

// ---------------------------------------------------------------------------
// Tiled bf16 WMMA GEMM: C[M,N] = A[M,K] @ B[K,N]
//   256 threads = 8 waves; block tile 128x64; wave tile 32x32 (4 WMMAs/K-step).
//   A tile staged by async global->LDS b128 copies; B tile staged transposed
//   with b64 loads + pair-packed b32 LDS stores.
// MODE 0: store bf16 C
// MODE 1: store bf16 relu(C + bias)
// MODE 2: fp32 C_out += C + bias   (residual accumulate)
// ---------------------------------------------------------------------------
template <int MODE>
__global__ __launch_bounds__(256) void gemm_bf16_kernel(const __bf16* __restrict__ A,
                                                        const __bf16* __restrict__ Bm,
                                                        const float* __restrict__ bias,
                                                        void* __restrict__ Cout,
                                                        int M, int N, int K) {
  __shared__ __attribute__((aligned(16))) __bf16 As[128 * 32];   // row-major [m][k]
  __shared__ __attribute__((aligned(16))) __bf16 Bts[64 * 32];   // transposed  [n][k]
  const int tid = threadIdx.x;
  const int lane = tid & 31;
  const int wid = tid >> 5;
  const int wm = (wid & 3) * 32;
  const int wn = (wid >> 2) * 32;
  const int m0 = blockIdx.x * 128;
  const int n0 = blockIdx.y * 64;

  f32x8 acc[4];
#pragma unroll
  for (int t = 0; t < 4; t++)
#pragma unroll
    for (int e = 0; e < 8; e++) acc[t][e] = 0.0f;

  for (int k0 = 0; k0 < K; k0 += 32) {
    // --- Stage A tile 128x32 via async global->LDS (2 x b128 per thread) ---
#pragma unroll
    for (int i = 0; i < 2; i++) {
      const int idx = (i * 256 + tid) * 8;       // element index, 8 bf16 per b128
      const int r = idx >> 5, c = idx & 31;      // c is a multiple of 8
      async_copy_b128(A + (size_t)(m0 + r) * K + k0 + c, &As[idx]);
    }
    // --- Stage B tile 32x64 transposed: thread covers k-pair x 4 n's ---
    {
      const int kk = (tid & 15) * 2;
      const int nn = (tid >> 4) * 4;
      union { uint2 p; __bf16 h[4]; } r0, r1;
      r0.p = *reinterpret_cast<const uint2*>(Bm + (size_t)(k0 + kk) * N + n0 + nn);
      r1.p = *reinterpret_cast<const uint2*>(Bm + (size_t)(k0 + kk + 1) * N + n0 + nn);
#pragma unroll
      for (int e = 0; e < 4; e++) {
        union { unsigned u; __bf16 h[2]; } pk;
        pk.h[0] = r0.h[e];
        pk.h[1] = r1.h[e];
        *reinterpret_cast<unsigned*>(&Bts[(nn + e) * 32 + kk]) = pk.u;
      }
    }
    wait_async0();
    __syncthreads();

    BfFrag a0, a1, b0, b1;
    load_a_frag(a0, &As[wm * 32], lane);
    load_a_frag(a1, &As[(wm + 16) * 32], lane);
    load_b_frag(b0, &Bts[wn * 32], lane);
    load_b_frag(b1, &Bts[(wn + 16) * 32], lane);
    acc[0] = wmma_bf16(a0, b0, acc[0]);
    acc[1] = wmma_bf16(a0, b1, acc[1]);
    acc[2] = wmma_bf16(a1, b0, acc[2]);
    acc[3] = wmma_bf16(a1, b1, acc[3]);
    __syncthreads();
  }

  // Epilogue: C layout -> vgpr r holds rows r / r+8 per 16-lane half, col = lane&15.
  const int colbase = n0 + wn + (lane & 15);
  const int rowhalf = (lane & 16) ? 8 : 0;
#pragma unroll
  for (int t = 0; t < 4; t++) {
    const int tm = (t >> 1) * 16, tn = (t & 1) * 16;
    const int col = colbase + tn;
#pragma unroll
    for (int r = 0; r < 8; r++) {
      const int row = m0 + wm + tm + rowhalf + r;
      float v = acc[t][r];
      if (MODE == 0) {
        ((__bf16*)Cout)[(size_t)row * N + col] = f2bf(v);
      } else if (MODE == 1) {
        v += bias[col];
        v = v > 0.0f ? v : 0.0f;
        ((__bf16*)Cout)[(size_t)row * N + col] = f2bf(v);
      } else {
        float* C = (float*)Cout;
        const size_t off = (size_t)row * N + col;
        C[off] = C[off] + v + bias[col];
      }
    }
  }
}

// ---------------------------------------------------------------------------
// Flash attention (unmasked, scores MULTIPLIED by sqrt(HD)=8 per reference bug).
// Q/K/V are bf16 [B*T, E] (head h occupies cols h*64..h*64+63).
// One wave per (b, h, 16-query tile); x += softmax(Q Kt * 8) V
// ---------------------------------------------------------------------------
__global__ __launch_bounds__(128) void flash_attn_kernel(const __bf16* __restrict__ Q,
                                                         const __bf16* __restrict__ Kc,
                                                         const __bf16* __restrict__ Vc,
                                                         float* __restrict__ X) {
  __shared__ __attribute__((aligned(16))) __bf16 ldsP[4][16 * 32];   // P tile row-major
  __shared__ __attribute__((aligned(16))) __bf16 ldsVt[4][64 * 32];  // V tile transposed [d][k]
  const int lane = threadIdx.x & 31;
  const int wid = threadIdx.x >> 5;
  const int gw = blockIdx.x * 4 + wid;
  const int qt = gw & 63;            // T/16 = 64 query tiles
  const int h = (gw >> 6) & 3;
  const int b = gw >> 8;
  const size_t E = 256;

  // Q A-fragments (d 0..31 and 32..63) held in VGPRs for the whole KV loop.
  BfFrag qa[2];
  {
    const int row = lane & 15;
    const int kb = (lane & 16) ? 8 : 0;
    const __bf16* p = Q + ((size_t)(b * 1024 + qt * 16 + row)) * E + h * 64;
#pragma unroll
    for (int df = 0; df < 2; df++)
#pragma unroll
      for (int i = 0; i < 8; i++) {
        const int d = df * 32 + kb + (i < 4 ? 2 * i : 16 + 2 * (i - 4));
        qa[df].u[i] = *reinterpret_cast<const unsigned int*>(p + d);
      }
  }

  float m[8], lsum[8];
#pragma unroll
  for (int r = 0; r < 8; r++) { m[r] = -1e30f; lsum[r] = 0.0f; }
  f32x8 o[4];
#pragma unroll
  for (int f = 0; f < 4; f++)
#pragma unroll
    for (int e = 0; e < 8; e++) o[f][e] = 0.0f;

  __bf16* Pw = ldsP[wid];
  __bf16* Vt = ldsVt[wid];

  for (int j = 0; j < 1024; j += 32) {
    // Stage V tile (32 keys x 64 d) transposed into LDS: Vt[d][k].
    // Wide b128 loads along d; transpose done by pair-packed b32 LDS stores.
    const __bf16* vbase = Vc + ((size_t)(b * 1024 + j)) * E + h * 64;
#pragma unroll
    for (int rep = 0; rep < 4; rep++) {
      const int kk = rep * 8 + (lane >> 3) * 2;     // adjacent key pair
      const int dbase = (lane & 7) * 8;
      union { uint4 p; __bf16 h[8]; } t0, t1;
      t0.p = *reinterpret_cast<const uint4*>(vbase + (size_t)kk * E + dbase);
      t1.p = *reinterpret_cast<const uint4*>(vbase + (size_t)(kk + 1) * E + dbase);
#pragma unroll
      for (int e = 0; e < 8; e++) {
        union { unsigned u; __bf16 h[2]; } pk;
        pk.h[0] = t0.h[e];
        pk.h[1] = t1.h[e];
        *reinterpret_cast<unsigned*>(&Vt[(dbase + e) * 32 + kk]) = pk.u;
      }
    }

    // S = Q @ K^T for keys j..j+31 : two 16x16 fragments, 2 WMMAs each (d split)
    f32x8 s[2];
#pragma unroll
    for (int half = 0; half < 2; half++) {
#pragma unroll
      for (int e = 0; e < 8; e++) s[half][e] = 0.0f;
      const int col = lane & 15;
      const int kb = (lane & 16) ? 8 : 0;
      const __bf16* kp = Kc + ((size_t)(b * 1024 + j + half * 16 + col)) * E + h * 64;
#pragma unroll
      for (int df = 0; df < 2; df++) {
        BfFrag kf;
#pragma unroll
        for (int i = 0; i < 8; i++)
          kf.u[i] = *reinterpret_cast<const unsigned int*>(kp + df * 32 + kb + (i < 4 ? 2 * i : 16 + 2 * (i - 4)));
        s[half] = wmma_bf16(qa[df], kf, s[half]);
      }
    }

    // Online softmax: each vgpr slot r is one row, spread over a 16-lane half.
    float p0[8], p1[8];
#pragma unroll
    for (int r = 0; r < 8; r++) {
      const float s0 = s[0][r] * 8.0f;        // faithful: scores *= sqrt(HD)
      const float s1 = s[1][r] * 8.0f;
      float mx = fmaxf(s0, s1);
#pragma unroll
      for (int msk = 8; msk >= 1; msk >>= 1) mx = fmaxf(mx, __shfl_xor(mx, msk, 32));
      const float mnew = fmaxf(m[r], mx);
      const float alpha = __expf(m[r] - mnew);
      const float e0 = __expf(s0 - mnew);
      const float e1 = __expf(s1 - mnew);
      float rs = e0 + e1;
#pragma unroll
      for (int msk = 8; msk >= 1; msk >>= 1) rs += __shfl_xor(rs, msk, 32);
      lsum[r] = lsum[r] * alpha + rs;
      m[r] = mnew;
      p0[r] = e0;
      p1[r] = e1;
#pragma unroll
      for (int f = 0; f < 4; f++) o[f][r] *= alpha;
    }

    // C-layout -> A-layout via LDS: store P row-major [16][32] as bf16
    {
      const int colA = lane & 15;
      const int rowb = (lane & 16) ? 8 : 0;
#pragma unroll
      for (int r = 0; r < 8; r++) {
        Pw[(rowb + r) * 32 + colA] = f2bf(p0[r]);
        Pw[(rowb + r) * 32 + 16 + colA] = f2bf(p1[r]);
      }
    }
    asm volatile("s_wait_dscnt 0" ::: "memory");

    // O += P @ V : one A-frag of P, four B-frags of V (d 0..15,16..31,32..47,48..63)
    BfFrag pa;
    load_a_frag(pa, Pw, lane);
#pragma unroll
    for (int f = 0; f < 4; f++) {
      BfFrag vf;
      load_b_frag(vf, &Vt[(f * 16) * 32], lane);
      o[f] = wmma_bf16(pa, vf, o[f]);
    }
  }

  // Normalize and accumulate into x (disjoint rows/cols per wave -> plain RMW)
  const int colA = lane & 15;
  const int rowb = (lane & 16) ? 8 : 0;
#pragma unroll
  for (int r = 0; r < 8; r++) {
    const float inv = 1.0f / lsum[r];
    const size_t xrow = ((size_t)(b * 1024 + qt * 16 + rowb + r)) * E + h * 64;
#pragma unroll
    for (int f = 0; f < 4; f++) {
      const size_t off = xrow + f * 16 + colA;
      X[off] += o[f][r] * inv;
    }
  }
}

// ---------------------------------------------------------------------------
// Host orchestration
// ---------------------------------------------------------------------------
extern "C" void kernel_launch(void* const* d_in, const int* in_sizes, int n_in,
                              void* d_out, int out_size, void* d_ws, size_t ws_size,
                              hipStream_t stream) {
  constexpr int Bv = 16, Tv = 1024, Ev = 256, Lv = 4;
  constexpr int M = Bv * Tv;             // 16384 tokens
  constexpr size_t MB = 1024ull * 1024ull;

  const int*   idx   = (const int*)d_in[0];
  const float* emb   = (const float*)d_in[1];
  const float* pos   = (const float*)d_in[2];
  const float* Wq    = (const float*)d_in[3];
  const float* Wk    = (const float*)d_in[4];
  const float* Wv    = (const float*)d_in[5];
  const float* ln1w  = (const float*)d_in[6];
  const float* ln1b  = (const float*)d_in[7];
  const float* ln2w  = (const float*)d_in[8];
  const float* ln2b  = (const float*)d_in[9];
  const float* W1    = (const float*)d_in[10];
  const float* b1    = (const float*)d_in[11];
  const float* W2    = (const float*)d_in[12];
  const float* b2    = (const float*)d_in[13];
  const float* lnfw  = (const float*)d_in[14];
  const float* lnfb  = (const float*)d_in[15];
  (void)in_sizes; (void)n_in; (void)out_size;

  // Workspace layout (~86 MB)
  if (ws_size < 88 * MB) return;
  char* ws = (char*)d_ws;
  float*  x    = (float*)(ws + 0);            // 16 MB  fp32 [M,E]
  __bf16* hbf  = (__bf16*)(ws + 16 * MB);     //  8 MB  bf16 [M,E]
  __bf16* qbf  = (__bf16*)(ws + 24 * MB);     //  8 MB
  __bf16* kbf  = (__bf16*)(ws + 32 * MB);     //  8 MB
  __bf16* vbf  = (__bf16*)(ws + 40 * MB);     //  8 MB
  __bf16* h1bf = (__bf16*)(ws + 48 * MB);     // 32 MB  bf16 [M,4E]
  __bf16* wqbf = (__bf16*)(ws + 80 * MB);     // 512 KB each for Wq/Wk/Wv
  __bf16* wkbf = wqbf + (size_t)Lv * Ev * Ev;
  __bf16* wvbf = wkbf + (size_t)Lv * Ev * Ev;
  __bf16* w1bf = wvbf + (size_t)Lv * Ev * Ev;         // 2 MB
  __bf16* w2bf = w1bf + (size_t)Lv * Ev * 4 * Ev;     // 2 MB

  // Weight conversion fp32 -> bf16
  {
    const int nEE = Lv * Ev * Ev;           // 262144
    const int nE4 = Lv * Ev * 4 * Ev;       // 1048576
    f32_to_bf16_kernel<<<nEE / 256, 256, 0, stream>>>(Wq, wqbf, nEE);
    f32_to_bf16_kernel<<<nEE / 256, 256, 0, stream>>>(Wk, wkbf, nEE);
    f32_to_bf16_kernel<<<nEE / 256, 256, 0, stream>>>(Wv, wvbf, nEE);
    f32_to_bf16_kernel<<<nE4 / 256, 256, 0, stream>>>(W1, w1bf, nE4);
    f32_to_bf16_kernel<<<nE4 / 256, 256, 0, stream>>>(W2, w2bf, nE4);
  }

  // Embedding + positional encoding
  embed_kernel<<<M, 256, 0, stream>>>(idx, emb, pos, x);

  const dim3 gEE(M / 128, Ev / 64);        // (128, 4)
  const dim3 gE4(M / 128, 4 * Ev / 64);    // (128, 16)

  for (int l = 0; l < Lv; l++) {
    // h = LN1(x) -> bf16
    layernorm_kernel<true><<<M, 256, 0, stream>>>(x, ln1w + l * Ev, ln1b + l * Ev, hbf);
    // Q, K, V projections
    gemm_bf16_kernel<0><<<gEE, 256, 0, stream>>>(hbf, wqbf + (size_t)l * Ev * Ev, nullptr,
                                                 qbf, M, Ev, Ev);
    gemm_bf16_kernel<0><<<gEE, 256, 0, stream>>>(hbf, wkbf + (size_t)l * Ev * Ev, nullptr,
                                                 kbf, M, Ev, Ev);
    gemm_bf16_kernel<0><<<gEE, 256, 0, stream>>>(hbf, wvbf + (size_t)l * Ev * Ev, nullptr,
                                                 vbf, M, Ev, Ev);
    // x += attention(Q,K,V)   (B*H*T/16 = 4096 waves, 4 waves/block)
    flash_attn_kernel<<<1024, 128, 0, stream>>>(qbf, kbf, vbf, x);
    // h2 = LN2(x) -> bf16
    layernorm_kernel<true><<<M, 256, 0, stream>>>(x, ln2w + l * Ev, ln2b + l * Ev, hbf);
    // h1 = relu(h2 @ W1 + b1) -> bf16
    gemm_bf16_kernel<1><<<gE4, 256, 0, stream>>>(hbf, w1bf + (size_t)l * Ev * 4 * Ev,
                                                 b1 + (size_t)l * 4 * Ev, h1bf, M, 4 * Ev, Ev);
    // x += h1 @ W2 + b2
    gemm_bf16_kernel<2><<<gEE, 256, 0, stream>>>(h1bf, w2bf + (size_t)l * Ev * 4 * Ev,
                                                 b2 + (size_t)l * Ev, x, M, Ev, 4 * Ev);
  }

  // Final LN -> fp32 output
  layernorm_kernel<false><<<M, 256, 0, stream>>>(x, lnfw, lnfb, d_out);
}
